// longformer_MHA_14843406975617
// MI455X (gfx1250) — matrix operand order repaired and checked by reading, not compiled, not run on hardware
//
#include <hip/hip_runtime.h>
#include <hip/hip_bf16.h>

// MI455X / gfx1250 (CDNA5): wave32, WMMA 16x16x32 bf16 -> f32, async LDS loads.
typedef __attribute__((ext_vector_type(16))) __bf16       v16bf;
typedef __attribute__((ext_vector_type(8)))  float        v8f;
typedef __attribute__((ext_vector_type(4)))  float        f32x4;
typedef __attribute__((ext_vector_type(4)))  unsigned int u32x4;

#define BB   2
#define SS   8192
#define DD   512
#define HH   8
#define WW   256
#define HDIM 64
#define CC   32
#define BHN  (BB*HH)
#define VSTR (3*WW + 8)   // 776 bf16 per row of transposed V3 tile (16B-aligned stride)

// LDS carve-up for band_attn (bytes)
#define LVT_BYTES ((size_t)HDIM * VSTR * 2)        // 99328: V3 transposed [d][key]
#define LK_BYTES  ((size_t)3 * WW * HDIM * 2)      // 98304: K3 row-major  [key][d]
#define LP_BYTES  ((size_t)8 * 16 * 32 * 4)        // 16384: per-wave P staging
#define SMEM_BYTES (LVT_BYTES + LK_BYTES + LP_BYTES)

union FragBF {
    u32x4 u[2];
    v16bf v;
};

// ---------------------------------------------------------------------------
// Kernel 1: fused QKV projection GEMM, bf16 WMMA, f32 accumulate.
// C = (X * W + bias) * scale, written bf16 to workspace in (B,H,S,HD) layout.
// Block: 256 threads (8 waves), 128x128 output tile, K-step 32.
// ---------------------------------------------------------------------------
__global__ __launch_bounds__(256) void qkv_gemm_bf16(
    const float* __restrict__ X, const float* __restrict__ Wm,
    const float* __restrict__ bias, __bf16* __restrict__ Out, float scale)
{
    __shared__ __bf16 lA[128][40];   // A tile [m][k], 16B-aligned row stride
    __shared__ __bf16 lB[128][40];   // B tile transposed [n][k]

    const int tid  = threadIdx.x;
    const int lane = tid & 31;
    const int wv   = tid >> 5;
    const int wm   = wv & 3;          // wave M position (4 waves along M)
    const int wn   = wv >> 2;         // wave N position (2 waves along N)
    const int m0   = blockIdx.x * 128;
    const int n0   = blockIdx.y * 128;

    const int fr = lane & 15;         // fragment row/col within 16
    const int kh = (lane >> 4) * 8;   // A-layout K half offset
    const int kb = (lane >> 4) * 16;  // contiguous-order K half offset (unused)

    v8f acc[2][4];
    #pragma unroll
    for (int i = 0; i < 2; i++)
        #pragma unroll
        for (int j = 0; j < 4; j++)
            acc[i][j] = (v8f){0,0,0,0,0,0,0,0};

    const int arow = tid >> 1;            // 0..127
    const int akh  = (tid & 1) * 16;      // 0 or 16
    const int brow = tid >> 3;            // 0..31 (k)
    const int bnb  = (tid & 7) * 16;      // 0..112 (n)

    for (int k0 = 0; k0 < DD; k0 += 32) {
        // ---- cooperative load A tile (128x32 f32 -> bf16, row major) ----
        {
            const float* src = X + (size_t)(m0 + arow) * DD + k0 + akh;
            #pragma unroll
            for (int i = 0; i < 4; i++) {
                f32x4 f = *(const f32x4*)(src + i * 4);
                #pragma unroll
                for (int j = 0; j < 4; j++)
                    lA[arow][akh + i * 4 + j] = (__bf16)f[j];
            }
        }
        // ---- cooperative load B tile (32x128 f32 -> bf16, transposed) ----
        {
            const float* src = Wm + (size_t)(k0 + brow) * DD + n0 + bnb;
            #pragma unroll
            for (int i = 0; i < 4; i++) {
                f32x4 f = *(const f32x4*)(src + i * 4);
                #pragma unroll
                for (int j = 0; j < 4; j++)
                    lB[bnb + i * 4 + j][brow] = (__bf16)f[j];
            }
        }
        __syncthreads();

        // ---- fragments: contiguous 16B LDS reads matching ISA layouts ----
        FragBF a[2], b[4];
        #pragma unroll
        for (int mt = 0; mt < 2; mt++) {
            const __bf16* p = &lA[wm * 32 + mt * 16 + fr][0];
            a[mt].u[0] = *(const u32x4*)(p + kh);
            a[mt].u[1] = *(const u32x4*)(p + kh + 16);
        }
        #pragma unroll
        for (int nt = 0; nt < 4; nt++) {
            const __bf16* p = &lB[wn * 64 + nt * 16 + fr][0];
            b[nt].u[0] = *(const u32x4*)(p + kh);
            b[nt].u[1] = *(const u32x4*)(p + kh + 16);
        }
        #pragma unroll
        for (int mt = 0; mt < 2; mt++)
            #pragma unroll
            for (int nt = 0; nt < 4; nt++)
                acc[mt][nt] = __builtin_amdgcn_wmma_f32_16x16x32_bf16(
                    false, a[mt].v, false, b[nt].v,
                    (short)0, acc[mt][nt], false, false);

        __syncthreads();
    }

    // ---- epilogue: bias + scale, scatter to (B,H,S,HD) bf16 ----
    const int rhalf = (lane >> 4) * 8;
    #pragma unroll
    for (int mt = 0; mt < 2; mt++) {
        #pragma unroll
        for (int nt = 0; nt < 4; nt++) {
            const int n  = n0 + wn * 64 + nt * 16 + fr;
            const float bn = bias[n];
            const int h = n >> 6, d = n & (HDIM - 1);
            #pragma unroll
            for (int rr = 0; rr < 8; rr++) {
                const int m = m0 + wm * 32 + mt * 16 + rr + rhalf;
                const int bi = m >> 13;            // m / S
                const int s  = m & (SS - 1);
                const float val = (acc[mt][nt][rr] + bn) * scale;
                Out[((size_t)(bi * HH + h) * SS + s) * HDIM + d] = (__bf16)val;
            }
        }
    }
}

// ---------------------------------------------------------------------------
// Kernel 2: banded attention, flash-style online softmax, all matmuls WMMA.
// One block (256 thr = 8 waves) per (bh, chunk c, half); wave owns 16 query
// rows. K3 staged row-major in LDS via GLOBAL_LOAD_ASYNC_TO_LDS_B128
// (ASYNCcnt path; removes 8x redundant global K traffic); V3 staged
// transposed for the PV B-operand.
// ---------------------------------------------------------------------------
__global__ __launch_bounds__(256) void band_attn(
    const __bf16* __restrict__ Q, const __bf16* __restrict__ K,
    const __bf16* __restrict__ V, float* __restrict__ Out)
{
    extern __shared__ char smem[];
    __bf16* lVt = (__bf16*)smem;                       // [HDIM][VSTR], V3^T
    __bf16* lK  = (__bf16*)(smem + LVT_BYTES);         // [3W][HDIM],  K3 row-major
    float*  lP  = (float*)(smem + LVT_BYTES + LK_BYTES);

    const int tid    = threadIdx.x;
    const int bid    = blockIdx.x;
    const int hsel   = bid & 1;
    const int c      = (bid >> 1) & (CC - 1);
    const int bh     = bid >> 6;
    const int kpos0  = (c - 1) * WW;     // global position of key window start

    const u32x4 zero4 = {0, 0, 0, 0};

    // ---- stage K3 (768 x 64 bf16) row-major via async LDS loads ----------
    {
        const unsigned lK_base = (unsigned)(uintptr_t)(void*)lK;  // LDS byte offset
        for (int ch = tid; ch < (3 * WW * 8); ch += 256) {        // 16B chunks
            const int key  = ch >> 3;
            const int part = ch & 7;
            const int gk   = kpos0 + key;
            if (gk >= 0 && gk < SS) {
                const unsigned loff = lK_base + (unsigned)(key * 128 + part * 16);
                const unsigned long long ga =
                    (unsigned long long)(uintptr_t)(K + ((size_t)bh * SS + gk) * HDIM + part * 8);
                asm volatile("global_load_async_to_lds_b128 %0, %1, off"
                             :: "v"(loff), "v"(ga) : "memory");
            } else {
                *(u32x4*)(lK + (size_t)key * HDIM + part * 8) = zero4;  // zero pad
            }
        }
    }

    // ---- stage V3 (768 x 64) transposed into LDS, zero-pad out-of-range --
    // (overlaps with the in-flight async K3 transfers)
    for (int key = tid; key < 3 * WW; key += 256) {
        const int gk = kpos0 + key;
        if (gk >= 0 && gk < SS) {
            const __bf16* src = V + ((size_t)bh * SS + gk) * HDIM;
            #pragma unroll
            for (int d8 = 0; d8 < HDIM; d8 += 8) {
                u32x4 raw = *(const u32x4*)(src + d8);
                const __bf16* e = (const __bf16*)&raw;
                #pragma unroll
                for (int j = 0; j < 8; j++)
                    lVt[(size_t)(d8 + j) * VSTR + key] = e[j];
            }
        } else {
            #pragma unroll
            for (int d = 0; d < HDIM; d++)
                lVt[(size_t)d * VSTR + key] = (__bf16)0.0f;
        }
    }

    asm volatile("s_wait_asynccnt 0" ::: "memory");  // async K3 done (this wave)
    __syncthreads();                                  // all waves' staging visible

    const int lane = tid & 31;
    const int wv   = tid >> 5;
    const int m0   = hsel * 128 + wv * 16;   // wave's query-row base in chunk
    const int fr   = lane & 15;
    const int kh   = (lane >> 4) * 8;        // A-layout K half
    const int kb   = (lane >> 4) * 16;       // contiguous-order K half (unused)
    float* lPw = lP + wv * (16 * 32);

    // ---- Q A-fragments (K = head dim, two 32-wide slices) ----
    FragBF aq[2];
    {
        const __bf16* qrow = Q + ((size_t)bh * SS + c * WW + m0 + fr) * HDIM;
        #pragma unroll
        for (int ks = 0; ks < 2; ks++) {
            aq[ks].u[0] = *(const u32x4*)(qrow + ks * 32 + kh);
            aq[ks].u[1] = *(const u32x4*)(qrow + ks * 32 + kh + 16);
        }
    }

    v8f o[4];
    #pragma unroll
    for (int nt = 0; nt < 4; nt++) o[nt] = (v8f){0,0,0,0,0,0,0,0};
    float mrow[8], lrow[8];
    #pragma unroll
    for (int rr = 0; rr < 8; rr++) { mrow[rr] = -1e30f; lrow[rr] = 0.0f; }

    const int pq_base = c * WW + m0;

    for (int jj = 0; jj < 24; jj++) {           // 24 steps of 32 keys = 768
        const int n0  = jj * 32;
        const int pk0 = kpos0 + n0 + fr;        // this lane's key col, tile 0
        const int pk1 = pk0 + 16;               // tile 1

        // ---- K B-fragments from LDS (contiguous head dim; zero-padded) ----
        FragBF bk0[2], bk1[2];
        #pragma unroll
        for (int ks = 0; ks < 2; ks++) {
            const __bf16* k0p = lK + (size_t)(n0 + fr) * HDIM + ks * 32 + kh;
            const __bf16* k1p = k0p + (size_t)16 * HDIM;
            bk0[ks].u[0] = *(const u32x4*)k0p;
            bk0[ks].u[1] = *(const u32x4*)(k0p + 16);
            bk1[ks].u[0] = *(const u32x4*)k1p;
            bk1[ks].u[1] = *(const u32x4*)(k1p + 16);
        }

        // ---- scores: two 16x16 tiles, K=64 over head dim ----
        v8f s0 = (v8f){0,0,0,0,0,0,0,0};
        v8f s1 = (v8f){0,0,0,0,0,0,0,0};
        s0 = __builtin_amdgcn_wmma_f32_16x16x32_bf16(false, aq[0].v, false, bk0[0].v, (short)0, s0, false, false);
        s0 = __builtin_amdgcn_wmma_f32_16x16x32_bf16(false, aq[1].v, false, bk0[1].v, (short)0, s0, false, false);
        s1 = __builtin_amdgcn_wmma_f32_16x16x32_bf16(false, aq[0].v, false, bk1[0].v, (short)0, s1, false, false);
        s1 = __builtin_amdgcn_wmma_f32_16x16x32_bf16(false, aq[1].v, false, bk1[1].v, (short)0, s1, false, false);

        // ---- band mask + online softmax (row reductions via 16-lane shfl) ----
        #pragma unroll
        for (int rr = 0; rr < 8; rr++) {
            const int row = rr + ((lane >> 4) << 3);
            const int pq  = pq_base + row;
            const bool v0 = (pk0 >= 0) && (pk0 < SS) && (pk0 >= pq - WW) && (pk0 <= pq + WW);
            const bool v1 = (pk1 >= 0) && (pk1 < SS) && (pk1 >= pq - WW) && (pk1 <= pq + WW);
            const float e0 = v0 ? s0[rr] : -1e30f;
            const float e1 = v1 ? s1[rr] : -1e30f;
            float t = fmaxf(e0, e1);
            t = fmaxf(t, __shfl_xor(t, 1, 32));
            t = fmaxf(t, __shfl_xor(t, 2, 32));
            t = fmaxf(t, __shfl_xor(t, 4, 32));
            t = fmaxf(t, __shfl_xor(t, 8, 32));
            const float mnew  = fmaxf(mrow[rr], t);
            const float alpha = __expf(mrow[rr] - mnew);
            mrow[rr] = mnew;
            const float p0 = v0 ? __expf(e0 - mnew) : 0.0f;
            const float p1 = v1 ? __expf(e1 - mnew) : 0.0f;
            float rs = p0 + p1;
            rs += __shfl_xor(rs, 1, 32);
            rs += __shfl_xor(rs, 2, 32);
            rs += __shfl_xor(rs, 4, 32);
            rs += __shfl_xor(rs, 8, 32);
            lrow[rr] = lrow[rr] * alpha + rs;
            #pragma unroll
            for (int nt = 0; nt < 4; nt++) o[nt][rr] *= alpha;
            lPw[row * 32 + fr]      = p0;   // stage P (C/D layout -> LDS)
            lPw[row * 32 + 16 + fr] = p1;
        }
        asm volatile("s_wait_dscnt 0" ::: "memory");

        // ---- reload P as bf16 A-fragment (16x32, K = keys) ----
        FragBF ap;
        {
            const float* pr = lPw + fr * 32;
            f32x4 f0 = *(const f32x4*)(pr + kh);
            f32x4 f1 = *(const f32x4*)(pr + kh + 4);
            f32x4 f2 = *(const f32x4*)(pr + 16 + kh);
            f32x4 f3 = *(const f32x4*)(pr + 16 + kh + 4);
            v16bf av;
            #pragma unroll
            for (int j = 0; j < 4; j++) {
                av[j]      = (__bf16)f0[j];
                av[4 + j]  = (__bf16)f1[j];
                av[8 + j]  = (__bf16)f2[j];
                av[12 + j] = (__bf16)f3[j];
            }
            ap.v = av;
        }

        // ---- accumulate O += P * V3 (B-fragments contiguous in LDS Vt) ----
        #pragma unroll
        for (int nt = 0; nt < 4; nt++) {
            const int dcol = nt * 16 + fr;
            const __bf16* vp = lVt + (size_t)dcol * VSTR + n0 + kh;
            FragBF bv;
            bv.u[0] = *(const u32x4*)vp;
            bv.u[1] = *(const u32x4*)(vp + 16);
            o[nt] = __builtin_amdgcn_wmma_f32_16x16x32_bf16(
                false, ap.v, false, bv.v, (short)0, o[nt], false, false);
        }
    }

    // ---- normalize and write output (B,S,D) f32 ----
    const int b = bh >> 3, h = bh & (HH - 1);
    #pragma unroll
    for (int nt = 0; nt < 4; nt++) {
        const int d = nt * 16 + fr;
        #pragma unroll
        for (int rr = 0; rr < 8; rr++) {
            const int row = rr + ((lane >> 4) << 3);
            const int s   = c * WW + m0 + row;
            const float val = o[nt][rr] / lrow[rr];
            Out[((size_t)b * SS + s) * DD + h * HDIM + d] = val;
        }
    }
}

// ---------------------------------------------------------------------------
extern "C" void kernel_launch(void* const* d_in, const int* in_sizes, int n_in,
                              void* d_out, int out_size, void* d_ws, size_t ws_size,
                              hipStream_t stream) {
    const float* x   = (const float*)d_in[0];
    const float* Wq  = (const float*)d_in[1];
    const float* bq  = (const float*)d_in[2];
    const float* Wk  = (const float*)d_in[3];
    const float* bkv = (const float*)d_in[4];
    const float* Wv  = (const float*)d_in[5];
    const float* bvv = (const float*)d_in[6];

    const size_t nElem = (size_t)BHN * SS * HDIM;   // 8,388,608 per tensor
    __bf16* qw = (__bf16*)d_ws;
    __bf16* kw = qw + nElem;
    __bf16* vw = kw + nElem;

    dim3 gg((BB * SS) / 128, DD / 128);             // 128 x 4 blocks
    qkv_gemm_bf16<<<gg, 256, 0, stream>>>(x, Wq, bq,  qw, 0.125f); // 1/sqrt(64)
    qkv_gemm_bf16<<<gg, 256, 0, stream>>>(x, Wk, bkv, kw, 1.0f);
    qkv_gemm_bf16<<<gg, 256, 0, stream>>>(x, Wv, bvv, vw, 1.0f);

    const int smem = (int)SMEM_BYTES;               // ~209 KB of 320 KB WGP LDS
    hipFuncSetAttribute(reinterpret_cast<const void*>(band_attn),
                        hipFuncAttributeMaxDynamicSharedMemorySize, smem);
    band_attn<<<dim3(BHN * CC * 2), 256, smem, stream>>>(qw, kw, vw, (float*)d_out);
}
